// WindowAttention3D_36472862278072
// MI455X (gfx1250) — matrix operand order repaired and checked by reading, not compiled, not run on hardware
//
#include <hip/hip_runtime.h>
#include <hip/hip_bf16.h>

typedef _Float16 v16h __attribute__((ext_vector_type(16)));
typedef _Float16 h8   __attribute__((ext_vector_type(8)));
typedef float    v8f  __attribute__((ext_vector_type(8)));
typedef int      v4i  __attribute__((ext_vector_type(4)));

#if __has_builtin(__builtin_amdgcn_global_load_async_to_lds_b128) && \
    __has_builtin(__builtin_amdgcn_s_wait_asynccnt)
#define HAVE_ASYNC_LDS 1
#else
#define HAVE_ASYNC_LDS 0
#endif

// 16-byte global -> LDS copy; async (ASYNCcnt-tracked) when available.
// Builtin signature: (int4 AS1*, int4 AS3*, imm offset, imm cpol).
__device__ __forceinline__ void copy_b128_to_lds(const _Float16* g,
                                                 _Float16* l) {
#if HAVE_ASYNC_LDS
  typedef __attribute__((address_space(1))) v4i gvec_t;
  typedef __attribute__((address_space(3))) v4i lvec_t;
  __builtin_amdgcn_global_load_async_to_lds_b128((gvec_t*)g, (lvec_t*)l, 0, 0);
#else
  *(uint4*)l = *(const uint4*)g;
#endif
}

__device__ __forceinline__ void wait_async_copies() {
#if HAVE_ASYNC_LDS
  __builtin_amdgcn_s_wait_asynccnt(0);
#endif
}

// ---------------------------------------------------------------------------
// WMMA fragment loaders per CDNA5 ISA 7.12.2 (wave32).
// A (16x32 f16, row-major, stride ld halves):
//   lane l: M = l&15, hi = l>>4
//   v16h elems 0..7  <- K = 8*hi .. 8*hi+7 ; elems 8..15 <- K = 16+8*hi ..
// B (32x16 f16, row-major KxN, stride ld halves):
//   lane l: N = l&15, hi = l>>4
//   v16h elems 0..7  <- K = 8*hi .. 8*hi+7 ; elems 8..15 <- K = 16+8*hi ..
// ---------------------------------------------------------------------------
__device__ __forceinline__ v16h load_a_frag(const _Float16* base, int ld) {
  int lane = threadIdx.x & 31;
  int m = lane & 15, hi = lane >> 4;
  const _Float16* p = base + m * ld + 8 * hi;
  h8 c0 = *(const h8*)(p);
  h8 c1 = *(const h8*)(p + 16);
  v16h a;
#pragma unroll
  for (int i = 0; i < 8; ++i) { a[i] = c0[i]; a[i + 8] = c1[i]; }
  return a;
}

__device__ __forceinline__ v16h load_b_frag(const _Float16* base, int ld) {
  int lane = threadIdx.x & 31;
  int n = lane & 15, hi = lane >> 4;
  const _Float16* p = base + 8 * hi * ld + n;
  v16h b;
#pragma unroll
  for (int i = 0; i < 8; ++i) { b[i] = p[i * ld]; b[i + 8] = p[(16 + i) * ld]; }
  return b;
}

__device__ __forceinline__ v8f wmma_f16(v16h a, v16h b, v8f c) {
  return __builtin_amdgcn_wmma_f32_16x16x32_f16(false, a, false, b,
                                                (short)0, c, false, false);
}

__device__ __forceinline__ float wave_max(float v) {
#pragma unroll
  for (int o = 16; o > 0; o >>= 1) v = fmaxf(v, __shfl_xor(v, o, 32));
  return v;
}
__device__ __forceinline__ float wave_sum(float v) {
#pragma unroll
  for (int o = 16; o > 0; o >>= 1) v += __shfl_xor(v, o, 32);
  return v;
}

#define WS_PER ((size_t)512 * 4 * 256 * 32)  // halves per q/k/v/o buffer

// ===========================================================================
// Kernel 1: QKV projection. out[m, n] = x[m,:] . qkv_w[n,:] + qkv_b[n]
// M = 131072, K = 128, Nout = 384. 64x64 tile per workgroup, 8 waves x 2
// WMMA tiles. Scatters into f16 workspaces (branch-free selects):
//   q: ws[0*PER + (bh*256+tok)*32 + d]   (pre-scaled by 1/sqrt(32))
//   k: ws[1*PER + (bh*32+d)*256 + tok]   (TRANSPOSED for attention B-frags)
//   v: ws[2*PER + (bh*256+tok)*32 + d]
// ===========================================================================
__global__ __launch_bounds__(256) void qkv_kernel(
    const float* __restrict__ x, const float* __restrict__ qkv_w,
    const float* __restrict__ qkv_b, _Float16* __restrict__ ws) {
  __shared__ _Float16 xs[64 * 128];  // A tile, row-major MxK
  __shared__ _Float16 wt[128 * 64];  // B tile, row-major KxN (transposed W)

  int tid = threadIdx.x;
  int m0 = blockIdx.x * 64;
  int n0 = blockIdx.y * 64;

  {  // stage x tile (f32 -> f16) and w tile transposed
    int row = tid >> 2;       // 0..63
    int kb = (tid & 3) * 32;  // 0,32,64,96
    const float* xsrc = x + (size_t)(m0 + row) * 128 + kb;
    const float* wsrc = qkv_w + (size_t)(n0 + row) * 128 + kb;
#pragma unroll
    for (int j = 0; j < 32; j += 4) {
      float4 fx = *(const float4*)(xsrc + j);
      xs[row * 128 + kb + j + 0] = (_Float16)fx.x;
      xs[row * 128 + kb + j + 1] = (_Float16)fx.y;
      xs[row * 128 + kb + j + 2] = (_Float16)fx.z;
      xs[row * 128 + kb + j + 3] = (_Float16)fx.w;
      float4 fw = *(const float4*)(wsrc + j);
      wt[(kb + j + 0) * 64 + row] = (_Float16)fw.x;
      wt[(kb + j + 1) * 64 + row] = (_Float16)fw.y;
      wt[(kb + j + 2) * 64 + row] = (_Float16)fw.z;
      wt[(kb + j + 3) * 64 + row] = (_Float16)fw.w;
    }
  }
  __syncthreads();

  int w = tid >> 5, lane = tid & 31, hi = lane >> 4, nl = lane & 15;
  int mt = w & 3;
  int ntb = (w >> 2) * 2;

  v8f acc[2] = {};
#pragma unroll
  for (int kk = 0; kk < 4; ++kk) {
    v16h a = load_a_frag(xs + mt * 16 * 128 + kk * 32, 128);
#pragma unroll
    for (int t = 0; t < 2; ++t) {
      v16h b = load_b_frag(wt + kk * 32 * 64 + (ntb + t) * 16, 64);
      acc[t] = wmma_f16(a, b, acc[t]);
    }
  }

  const float scale = 0.17677669529663687f;  // 1/sqrt(32)
#pragma unroll
  for (int t = 0; t < 2; ++t) {
#pragma unroll
    for (int r = 0; r < 8; ++r) {
      int gm = m0 + mt * 16 + r + 8 * hi;
      int gn = n0 + (ntb + t) * 16 + nl;
      float v = acc[t][r] + qkv_b[gn];
      int which = gn >> 7;  // 0=q,1=k,2=v
      int c = gn & 127;
      int h = c >> 5, d = c & 31;
      size_t bh = (size_t)((gm >> 8) * 4 + h);
      int tok = gm & 255;
      size_t qvidx = (bh * 256 + tok) * 32 + d;
      size_t kidx = (bh * 32 + d) * 256 + tok;
      size_t di = (size_t)which * WS_PER + (which == 1 ? kidx : qvidx);
      float mul = (which == 0) ? scale : 1.0f;
      ws[di] = (_Float16)(v * mul);
    }
  }
}

// ===========================================================================
// Kernel 2: windowed attention, one workgroup per (b,h). 2048 blocks.
// Q/Kt/V staged with ASYNC global->LDS copies (all straight copies now).
// S = QK^T (+bias+mask) -> softmax -> O = P V, in 64-row blocks.
// ===========================================================================
__global__ __launch_bounds__(256) void attn_kernel(
    const _Float16* __restrict__ ws, const float* __restrict__ mask,
    const float* __restrict__ bias_table, const int* __restrict__ rel_idx,
    _Float16* __restrict__ ows) {
  __shared__ _Float16 Qs[256 * 32];  // 16 KB, row-major [tok][d] (pre-scaled)
  __shared__ _Float16 Kt[32 * 256];  // 16 KB, transposed  [d][tok]
  __shared__ _Float16 Vs[256 * 32];  // 16 KB, row-major [tok][d]
  __shared__ float Sb[64 * 256];     // 64 KB score block
  __shared__ _Float16 Pb[64 * 256];  // 32 KB softmax probs (f16)

  int tid = threadIdx.x;
  int bh = blockIdx.x;
  int b = bh >> 2, h = bh & 3;
  size_t base = (size_t)bh * 256 * 32;

  {  // async-stage Q, Kt, V (each 8192 halves = 1024 x b128; 4 per thread)
    const _Float16* gq = ws + base;               // q buffer
    const _Float16* gk = ws + WS_PER + base;      // k (already transposed)
    const _Float16* gv = ws + 2 * WS_PER + base;  // v buffer
#pragma unroll
    for (int i = 0; i < 4; ++i) {
      int off = (tid + 256 * i) * 8;  // half offset of a 16B chunk
      copy_b128_to_lds(gq + off, Qs + off);
      copy_b128_to_lds(gk + off, Kt + off);
      copy_b128_to_lds(gv + off, Vs + off);
    }
    wait_async_copies();
  }
  __syncthreads();

  int w = tid >> 5, lane = tid & 31, hi = lane >> 4, nl = lane & 15;
  const float* mrow = mask + (size_t)(b & 63) * 65536;

  for (int mb = 0; mb < 4; ++mb) {
    // ----- S = Q K^T + bias + mask (4 m-tiles x 16 n-tiles = 64 WMMA) -----
    {
      int mt = w >> 1;
      v16h a = load_a_frag(Qs + (mb * 64 + mt * 16) * 32, 32);
#pragma unroll
      for (int i = 0; i < 8; ++i) {
        int nt = (w & 1) * 8 + i;
        v16h bf = load_b_frag(Kt + nt * 16, 256);
        v8f c = {};
        c = wmma_f16(a, bf, c);
#pragma unroll
        for (int r = 0; r < 8; ++r) {
          int ml = mt * 16 + r + 8 * hi;
          int gm = mb * 64 + ml;
          int gn = nt * 16 + nl;
          int ridx = rel_idx[gm * 256 + gn];
          float v = c[r] + bias_table[ridx * 4 + h] + mrow[gm * 256 + gn];
          Sb[ml * 256 + gn] = v;
        }
      }
    }
    __syncthreads();

    // ----- softmax over rows, write P (f16). 8 rows per wave. -----
    for (int rr = 0; rr < 8; ++rr) {
      int ml = w * 8 + rr;
      const float* rp = Sb + ml * 256 + lane * 8;
      float4 f0 = *(const float4*)rp;
      float4 f1 = *(const float4*)(rp + 4);
      float v0 = f0.x, v1 = f0.y, v2 = f0.z, v3 = f0.w;
      float v4 = f1.x, v5 = f1.y, v6 = f1.z, v7 = f1.w;
      float mx = fmaxf(fmaxf(fmaxf(v0, v1), fmaxf(v2, v3)),
                       fmaxf(fmaxf(v4, v5), fmaxf(v6, v7)));
      mx = wave_max(mx);
      float e0 = __expf(v0 - mx), e1 = __expf(v1 - mx);
      float e2 = __expf(v2 - mx), e3 = __expf(v3 - mx);
      float e4 = __expf(v4 - mx), e5 = __expf(v5 - mx);
      float e6 = __expf(v6 - mx), e7 = __expf(v7 - mx);
      float s = wave_sum(e0 + e1 + e2 + e3 + e4 + e5 + e6 + e7);
      float inv = 1.0f / s;
      _Float16* pp = Pb + ml * 256 + lane * 8;
      pp[0] = (_Float16)(e0 * inv); pp[1] = (_Float16)(e1 * inv);
      pp[2] = (_Float16)(e2 * inv); pp[3] = (_Float16)(e3 * inv);
      pp[4] = (_Float16)(e4 * inv); pp[5] = (_Float16)(e5 * inv);
      pp[6] = (_Float16)(e6 * inv); pp[7] = (_Float16)(e7 * inv);
    }
    __syncthreads();

    // ----- O block = P V  (4x2 tiles, one per wave, K=256 in 8 steps) -----
    {
      int mt = w >> 1, nt = w & 1;
      v8f o = {};
#pragma unroll
      for (int kk = 0; kk < 8; ++kk) {
        v16h a = load_a_frag(Pb + mt * 16 * 256 + kk * 32, 256);
        v16h bf = load_b_frag(Vs + kk * 32 * 32 + nt * 16, 32);
        o = wmma_f16(a, bf, o);
      }
#pragma unroll
      for (int r = 0; r < 8; ++r) {
        int tok = mb * 64 + mt * 16 + r + 8 * hi;
        int d = nt * 16 + nl;
        ows[((size_t)b * 256 + tok) * 128 + h * 32 + d] = (_Float16)o[r];
      }
    }
    __syncthreads();
  }
}

// ===========================================================================
// Kernel 3: output projection. out[m,n] = o[m,:] . proj_w[n,:] + proj_b[n]
// M = 131072, K = 128, N = 128. A tile staged with async copies; f32 out.
// ===========================================================================
__global__ __launch_bounds__(256) void proj_kernel(
    const _Float16* __restrict__ ows, const float* __restrict__ proj_w,
    const float* __restrict__ proj_b, float* __restrict__ out) {
  __shared__ _Float16 xs[64 * 128];
  __shared__ _Float16 wt[128 * 64];

  int tid = threadIdx.x;
  int m0 = blockIdx.x * 64;
  int n0 = blockIdx.y * 64;

  {  // async-stage A tile (already f16); stage transposed W tile (f32->f16)
    const _Float16* ag = ows + (size_t)m0 * 128;
#pragma unroll
    for (int i = 0; i < 4; ++i) {
      int off = (tid + 256 * i) * 8;
      copy_b128_to_lds(ag + off, xs + off);
    }
    int row = tid >> 2;
    int kb = (tid & 3) * 32;
    const float* wsrc = proj_w + (size_t)(n0 + row) * 128 + kb;
#pragma unroll
    for (int j = 0; j < 32; j += 4) {
      float4 fw = *(const float4*)(wsrc + j);
      wt[(kb + j + 0) * 64 + row] = (_Float16)fw.x;
      wt[(kb + j + 1) * 64 + row] = (_Float16)fw.y;
      wt[(kb + j + 2) * 64 + row] = (_Float16)fw.z;
      wt[(kb + j + 3) * 64 + row] = (_Float16)fw.w;
    }
    wait_async_copies();
  }
  __syncthreads();

  int w = tid >> 5, lane = tid & 31, hi = lane >> 4, nl = lane & 15;
  int mt = w & 3;
  int ntb = (w >> 2) * 2;

  v8f acc[2] = {};
#pragma unroll
  for (int kk = 0; kk < 4; ++kk) {
    v16h a = load_a_frag(xs + mt * 16 * 128 + kk * 32, 128);
#pragma unroll
    for (int t = 0; t < 2; ++t) {
      v16h b = load_b_frag(wt + kk * 32 * 64 + (ntb + t) * 16, 64);
      acc[t] = wmma_f16(a, b, acc[t]);
    }
  }
#pragma unroll
  for (int t = 0; t < 2; ++t) {
#pragma unroll
    for (int r = 0; r < 8; ++r) {
      int gm = m0 + mt * 16 + r + 8 * hi;
      int gn = n0 + (ntb + t) * 16 + nl;
      out[(size_t)gm * 128 + gn] = acc[t][r] + proj_b[gn];
    }
  }
}

// ===========================================================================
extern "C" void kernel_launch(void* const* d_in, const int* in_sizes, int n_in,
                              void* d_out, int out_size, void* d_ws,
                              size_t ws_size, hipStream_t stream) {
  const float* x = (const float*)d_in[0];
  const float* mask = (const float*)d_in[1];
  const float* qkv_w = (const float*)d_in[2];
  const float* qkv_b = (const float*)d_in[3];
  const float* proj_w = (const float*)d_in[4];
  const float* proj_b = (const float*)d_in[5];
  const float* bias_table = (const float*)d_in[6];
  const int* rel_idx = (const int*)d_in[7];

  _Float16* ws = (_Float16*)d_ws;
  _Float16* ows = ws + 3 * WS_PER;

  qkv_kernel<<<dim3(2048, 6), 256, 0, stream>>>(x, qkv_w, qkv_b, ws);
  attn_kernel<<<dim3(2048), 256, 0, stream>>>(ws, mask, bias_table, rel_idx,
                                              ows);
  proj_kernel<<<dim3(2048, 2), 256, 0, stream>>>(ows, proj_w, proj_b,
                                                 (float*)d_out);
}
